// HistLoss_64192581206125
// MI455X (gfx1250) — matrix operand order repaired and checked by reading, not compiled, not run on hardware
//
#include <hip/hip_runtime.h>
#include <math.h>

typedef __attribute__((ext_vector_type(16))) _Float16 v16h;
typedef __attribute__((ext_vector_type(8)))  float    v8f;

#define EPSF      1e-6f
#define SIG2      4.0e-4f          // sigma^2, sigma = 0.02
#define HB        128
#define NPIX      65536            // 256*256
#define NB        4
#define NGEMM     24               // 2 imgs * 4 batch * 3 channel-pairs
#define KSPLIT    32
#define KSLICE    (NPIX / KSPLIT)  // 2048
#define CHUNK     32
#define NCHUNK    (KSLICE / CHUNK) // 64
#define MU_STEP   (6.0f / 127.0f)

// workspace layout (float offsets)
#define OFF_IU    0
#define SZ_IU     (2 * NB * 3 * NPIX)
#define OFF_IV    (OFF_IU + SZ_IU)
#define OFF_J     (OFF_IV + SZ_IU)
#define SZ_J      (2 * NB * NPIX)
#define OFF_HIST  (OFF_J + SZ_J)
#define SZ_HIST   (NGEMM * HB * HB)
#define OFF_NORM  (OFF_HIST + SZ_HIST)
#define OFF_ACC   (OFF_NORM + 8)
#define SZ_ZERO   (SZ_HIST + 8 + 1)

__global__ void zero_kernel(float* p, int n) {
    int i = blockIdx.x * blockDim.x + threadIdx.x;
    if (i < n) p[i] = 0.0f;
}

// One thread per (img, b, pixel): compute Iy-scale J, and Iu/Iv for the 3 channel pairs.
__global__ void prep_kernel(const float* __restrict__ pred,
                            const float* __restrict__ target,
                            float* __restrict__ ws) {
    int tid = blockIdx.x * blockDim.x + threadIdx.x;   // exactly 2*4*65536 threads
    int img = tid >> 18;
    int b   = (tid >> 16) & 3;
    int p   = tid & (NPIX - 1);
    const float* src  = (img == 0) ? pred : target;
    const float* base = src + (size_t)b * 3 * NPIX;
    float r  = base[p];
    float g  = base[NPIX + p];
    float bl = base[2 * NPIX + p];
    r  = fminf(fmaxf(r , 0.f), 1.f);
    g  = fminf(fmaxf(g , 0.f), 1.f);
    bl = fminf(fmaxf(bl, 0.f), 1.f);
    float iy = sqrtf(r * r + g * g + bl * bl + EPSF);
    float lr = logf(r  + EPSF);
    float lg = logf(g  + EPSF);
    float lb = logf(bl + EPSF);

    float* Iu = ws + OFF_IU;
    float* Iv = ws + OFF_IV;
    float* Jw = ws + OFF_J;
    int ib = img * NB + b;
    size_t c0 = ((size_t)ib * 3 + 0) * NPIX + p;
    size_t c1 = ((size_t)ib * 3 + 1) * NPIX + p;
    size_t c2 = ((size_t)ib * 3 + 2) * NPIX + p;
    Iu[c0] = lr - lg;  Iv[c0] = lr - lb;   // r vs (g, b)
    Iu[c1] = lg - lr;  Iv[c1] = lg - lb;   // g vs (r, b)
    Iu[c2] = lb - lr;  Iv[c2] = lb - lg;   // b vs (r, g)
    Jw[(size_t)ib * NPIX + p] = SIG2 * iy; // fold sigma^2 * Iy into A numerator
}

// 8 waves per block. Wave w: u-rows [16w, 16w+16), all 8 v-tiles.
// A fragment (16x32 f16) built per documented layout: lane m=lane&15 (M), kg=lane>>4,
// half j -> K = (j<8 ? j : j+8) + 8*kg  (contiguous global loads).
// B fragment (32x16 f16): same K order as A (half j -> K = (j<8 ? j : j+8) + 8*kg),
// produced once per chunk by wave w for v-tile w, shared through LDS.
__global__ __launch_bounds__(256) void hist_gemm_kernel(const float* __restrict__ ws,
                                                        float* __restrict__ hist) {
    __shared__ v16h ldsB[8 * 32];   // 8 v-tiles * 32 lanes * 32B = 8 KB

    const float* Iu = ws + OFF_IU;
    const float* Iv = ws + OFF_IV;
    const float* Jw = ws + OFF_J;

    int g  = blockIdx.x % NGEMM;
    int ks = blockIdx.x / NGEMM;
    int img = g / (NB * 3);
    int b   = (g / 3) % NB;
    int ch  = g % 3;
    int ib  = img * NB + b;
    const float* IuG = Iu + ((size_t)ib * 3 + ch) * NPIX;
    const float* IvG = Iv + ((size_t)ib * 3 + ch) * NPIX;
    const float* JG  = Jw + (size_t)ib * NPIX;
    float* histG = hist + (size_t)g * HB * HB;

    int lane = threadIdx.x & 31;
    int w    = threadIdx.x >> 5;
    int kg   = lane >> 4;
    int m    = lane & 15;
    // mu for this lane's u-row; identical to mu for column n=16w+m of v-tile w.
    float mu_u = fmaf((float)(w * 16 + m), MU_STEP, -3.0f);

    v8f acc[8] = {};

    int kbase = ks * KSLICE;
    for (int c = 0; c < NCHUNK; ++c) {
        int p0 = kbase + c * CHUNK;

        // ---- produce B fragment for v-tile w (dv kernel values) ----
        const float* pv = IvG + p0 + kg * 8;   // K = (j<8?j:j+8)+8*kg
        v16h bf;
#pragma unroll
        for (int j = 0; j < 16; ++j) {
            float d = pv[(j < 8) ? j : (j + 8)] - mu_u;
            float t = fmaf(d, d, SIG2);
            bf[j] = (_Float16)(SIG2 * __builtin_amdgcn_rcpf(t));
        }

        // ---- build A fragment (J * du kernel values) ----
        const float* pa = IuG + p0 + kg * 8;
        const float* pj = JG  + p0 + kg * 8;
        v16h af;
#pragma unroll
        for (int j = 0; j < 16; ++j) {
            int o = (j < 8) ? j : (j + 8);     // K = (j<8?j:j+8)+8*kg
            float d = pa[o] - mu_u;
            float t = fmaf(d, d, SIG2);
            af[j] = (_Float16)(pj[o] * __builtin_amdgcn_rcpf(t));
        }

        ldsB[w * 32 + lane] = bf;
        __syncthreads();

#pragma unroll
        for (int t8 = 0; t8 < 8; ++t8) {
            v16h bt = ldsB[t8 * 32 + lane];
            acc[t8] = __builtin_amdgcn_wmma_f32_16x16x32_f16(
                false, af, false, bt, (short)0, acc[t8], false, false);
        }
        __syncthreads();   // consumers done before next chunk's producers overwrite
    }

    // C/D layout: VGPR r -> M = r + 8*kg, N = m. K-split partials merge via f32 atomics.
#pragma unroll
    for (int t8 = 0; t8 < 8; ++t8) {
#pragma unroll
        for (int r = 0; r < 8; ++r) {
            int u = w * 16 + kg * 8 + r;
            int v = t8 * 16 + m;
            atomicAdd(&histG[u * HB + v], acc[t8][r]);
        }
    }
}

// One block per (img, b): norm = sum over (ch,u,v) + EPS
__global__ void norm_kernel(const float* __restrict__ hist, float* __restrict__ norm) {
    __shared__ float red[256];
    const float* hb = hist + (size_t)blockIdx.x * 3 * HB * HB;
    float s = 0.0f;
    for (int i = threadIdx.x; i < 3 * HB * HB; i += 256) s += hb[i];
    red[threadIdx.x] = s;
    __syncthreads();
    for (int off = 128; off > 0; off >>= 1) {
        if (threadIdx.x < off) red[threadIdx.x] += red[threadIdx.x + off];
        __syncthreads();
    }
    if (threadIdx.x == 0) norm[blockIdx.x] = red[0] + EPSF;
}

// Hellinger: sum over all (b,ch,u,v) of (sqrt(t/tn) - sqrt(p/pn))^2
__global__ void loss_kernel(const float* __restrict__ hist,
                            const float* __restrict__ norm,
                            float* __restrict__ acc) {
    __shared__ float red[256];
    int tid = blockIdx.x * 256 + threadIdx.x;
    const int NTOT = NB * 3 * HB * HB;
    float s = 0.0f;
    for (int e = tid; e < NTOT; e += gridDim.x * 256) {
        int b   = e / (3 * HB * HB);
        int rem = e % (3 * HB * HB);
        float ph = hist[(size_t)b        * 3 * HB * HB + rem];
        float th = hist[(size_t)(NB + b) * 3 * HB * HB + rem];
        float d  = sqrtf(th / norm[NB + b]) - sqrtf(ph / norm[b]);
        s += d * d;
    }
    red[threadIdx.x] = s;
    __syncthreads();
    for (int off = 128; off > 0; off >>= 1) {
        if (threadIdx.x < off) red[threadIdx.x] += red[threadIdx.x + off];
        __syncthreads();
    }
    if (threadIdx.x == 0) atomicAdd(acc, red[0]);
}

__global__ void finalize_kernel(const float* __restrict__ acc, float* __restrict__ out) {
    out[0] = sqrtf(acc[0]) * 0.70710678118654752f * (1.0f / (float)NB);
}

extern "C" void kernel_launch(void* const* d_in, const int* in_sizes, int n_in,
                              void* d_out, int out_size, void* d_ws, size_t ws_size,
                              hipStream_t stream) {
    const float* pred   = (const float*)d_in[0];
    const float* target = (const float*)d_in[1];
    float* ws   = (float*)d_ws;
    float* out  = (float*)d_out;
    float* hist = ws + OFF_HIST;
    float* norm = ws + OFF_NORM;
    float* acc  = ws + OFF_ACC;

    zero_kernel<<<(SZ_ZERO + 255) / 256, 256, 0, stream>>>(hist, SZ_ZERO);
    prep_kernel<<<(2 * NB * NPIX) / 256, 256, 0, stream>>>(pred, target, ws);
    hist_gemm_kernel<<<NGEMM * KSPLIT, 256, 0, stream>>>(ws, hist);
    norm_kernel<<<8, 256, 0, stream>>>(hist, norm);
    loss_kernel<<<192, 256, 0, stream>>>(hist, norm, acc);
    finalize_kernel<<<1, 1, 0, stream>>>(acc, out);
}
